// sLSTM_43301860278400
// MI455X (gfx1250) — compile-verified, matrix operand order checked by reading, not run on hardware
//
#include <hip/hip_runtime.h>
#include <hip/hip_bf16.h>
#include <math.h>

// ---------------------------------------------------------------------------
// sLSTM (xLSTM exponential-gate) 2-layer recurrent kernel for MI455X (gfx1250)
//
// - bf16 WMMA 16x16x32, M = batch = 16 (perfect shape match)
// - weights converted once to bf16 (32MB, L2-resident)
// - x pre-converted to bf16 time-major [T][B][D]; h-state stored bf16
// - A-tiles staged into LDS by the Tensor Data Mover (tensor_load_to_lds +
//   s_wait_tensorcnt), double-buffered 16x512 chunks, straight-line pipeline
//   so weight streams stay on the GLOBAL (not FLAT) path
// - persistent 32-WG kernel, 2 sense-reversing global barriers per timestep
// - cell state kept in LDS for the whole sequence
// ---------------------------------------------------------------------------

typedef __attribute__((ext_vector_type(16))) __bf16 v16bf;
typedef __attribute__((ext_vector_type(8)))  float  v8f;
typedef __attribute__((ext_vector_type(4)))  unsigned int v4u;
typedef __attribute__((ext_vector_type(8)))  int    v8i;
typedef __attribute__((ext_vector_type(4)))  int    v4i;

#define BSZ     16
#define TT      1024
#define DD      1024
#define HH      1024
#define NWG     32
#define THREADS 256
#define WMAT    (4096u * 1024u)   // elements per weight matrix (4H x K) = 2^22
#define BH      (BSZ * HH)        // 16384
#define CHK     512               // K-chunk staged per TDM descriptor

#if __has_include(<hip/amd_detail/amd_gfx1250_TDM.h>)
#define TDM_6ARG 1
#endif

// ---------------- weight f32 -> bf16 conversion ----------------------------
__global__ __launch_bounds__(THREADS) void convert_weights(
    const float* __restrict__ a, const float* __restrict__ b,
    const float* __restrict__ c, const float* __restrict__ d,
    unsigned short* __restrict__ o_raw) {
  size_t i = (size_t)blockIdx.x * THREADS + threadIdx.x;   // 0 .. 4*WMAT-1
  size_t which = i >> 22;
  size_t off   = i & (WMAT - 1);
  const float* s = (which == 0) ? a : (which == 1) ? b : (which == 2) ? c : d;
  ((__bf16*)o_raw)[i] = (__bf16)s[off];
}

// ---------------- x f32 [B,T,D] -> bf16 time-major [T,B,D] -----------------
__global__ __launch_bounds__(THREADS) void convert_x(
    const float* __restrict__ x, unsigned short* __restrict__ xb_raw) {
  size_t i = (size_t)blockIdx.x * THREADS + threadIdx.x;   // 0 .. B*T*D-1
  size_t t = i >> 14;            // 16*1024 elements per timestep tile
  size_t r = i & 16383;
  size_t b = r >> 10;
  size_t d = r & 1023;
  ((__bf16*)xb_raw)[i] = (__bf16)x[(b * TT + t) * (size_t)DD + d];
}

// ---------------- zero barrier + bf16 h double-buffers ---------------------
__global__ __launch_bounds__(THREADS) void init_state(
    int* __restrict__ bar, unsigned short* __restrict__ hbufs, int n) {
  int i = blockIdx.x * THREADS + threadIdx.x;
  if (i < 2) bar[i] = 0;
  if (i < n) hbufs[i] = 0;
}

// ---------------- TDM: 2D tile (16 x CHK bf16) global -> LDS ---------------
// D# per CDNA5 ISA 8.3/8.4: group0 = {count/type, lds_addr, global_addr},
// group1 = {data_size, tensor dims, tile dims, dim0 stride}. Groups 2/3 zero
// (2D tensor: tile_dim2 = 0).
__device__ __forceinline__ void tdm_load_tile(
    unsigned lds_addr, const __bf16* __restrict__ gsrc,
    unsigned rows, unsigned cols, unsigned rowStrideElems) {
  unsigned long long ga = (unsigned long long)(uintptr_t)gsrc;
  v4u g0;
  g0.x = 1u;                                     // count=1, user descriptor
  g0.y = lds_addr;                               // LDS byte address
  g0.z = (unsigned)ga;                           // global_addr[31:0]
  g0.w = (unsigned)((ga >> 32) & 0x01FFFFFFu) | (2u << 30);  // [56:32]+type=2
  v8i g1;
  g1[0] = (int)(1u << 16);                       // data_size = 1 (2 bytes)
  g1[1] = (int)(cols << 16);                     // tensor_dim0[15:0]
  g1[2] = (int)(rows << 16);                     // dim0[31:16]=0 | tensor_dim1
  g1[3] = (int)(cols << 16);                     // dim1[31:16]=0 | tile_dim0
  g1[4] = (int)rows;                             // tile_dim1 | tile_dim2=0
  g1[5] = (int)rowStrideElems;                   // tensor_dim0_stride[31:0]
  g1[6] = 0;                                     // stride[47:32] | dim1_stride
  g1[7] = 0;
  v4i z4 = {0, 0, 0, 0};
#ifdef TDM_6ARG
  v8i z8 = {0, 0, 0, 0, 0, 0, 0, 0};
  __builtin_amdgcn_tensor_load_to_lds(g0, g1, z4, z4, z8, 0);
#else
  __builtin_amdgcn_tensor_load_to_lds(g0, g1, z4, z4, 0);
#endif
}

// ---------------- persistent recurrent kernel ------------------------------
__global__ __launch_bounds__(THREADS, 1) void slstm_persistent(
    const float* __restrict__ b0,                 // [4H]
    const float* __restrict__ b1,                 // [4H]
    const unsigned short* __restrict__ xb_raw,    // [T][B][D] bf16
    const unsigned short* __restrict__ wih0_raw,  // [4H,D] bf16
    const unsigned short* __restrict__ whh0_raw,  // [4H,H] bf16
    const unsigned short* __restrict__ wih1_raw,  // [4H,H] bf16
    const unsigned short* __restrict__ whh1_raw,  // [4H,H] bf16
    unsigned short* __restrict__ h0buf_raw,       // [2][B][H] bf16
    unsigned short* __restrict__ h1buf_raw,       // [2][B][H] bf16
    int*   __restrict__ bar,                      // [count, sense]
    float* __restrict__ out,                      // [B,T,H]
    float* __restrict__ h0f, float* __restrict__ c0f,
    float* __restrict__ h1f, float* __restrict__ c1f) {

  const __bf16* xb   = (const __bf16*)xb_raw;
  const __bf16* wih0 = (const __bf16*)wih0_raw;
  const __bf16* whh0 = (const __bf16*)whh0_raw;
  const __bf16* wih1 = (const __bf16*)wih1_raw;
  const __bf16* whh1 = (const __bf16*)whh1_raw;
  __bf16* h0buf = (__bf16*)h0buf_raw;
  __bf16* h1buf = (__bf16*)h1buf_raw;

  __shared__ __bf16 Abuf[2][16][CHK];   // 2 x 16KB double-buffered A chunks
  __shared__ float  gbuf[4][16][32];    // 8 KB staged gate pre-activations
  __shared__ float  c0s[16][32];        // cell state, layer 0
  __shared__ float  c1s[16][32];        // cell state, layer 1
  __shared__ int    lsense;

  const int lane  = threadIdx.x & 31;
  const int wave  = threadIdx.x >> 5;       // 8 waves
  const int gate  = wave >> 1;              // 0..3 -> i,f,g,o
  const int tile  = wave & 1;               // 2 column-tiles per gate
  const int mrow  = lane & 15;              // A-operand matrix row (batch)
  const int klane = (lane >> 4) << 4;       // 0 or 16: lane's K sub-block
  const int uBase = blockIdx.x * 32;        // hidden units owned by this WG
  const int nrow  = gate * 1024 + uBase + tile * 16 + mrow;  // W row

  const unsigned ldsA0 = (unsigned)(uintptr_t)&Abuf[0][0][0];
  const unsigned ldsA1 = (unsigned)(uintptr_t)&Abuf[1][0][0];

  if (threadIdx.x == 0) lsense = 0;
  for (int p = threadIdx.x; p < 512; p += THREADS) {
    c0s[p >> 5][p & 31] = 0.0f;
    c1s[p >> 5][p & 31] = 0.0f;
  }
  __syncthreads();

  // 16 WMMAs over one staged 16xCHK chunk; W flows straight from kernel args
  // so these stay global_load_b128 (LOADcnt only, decoupled from LDS waits).
  auto gemmChunk = [&](const __bf16* __restrict__ W, int ko, int buf,
                       v8f acc) -> v8f {
    const __bf16* wptr = W + (size_t)nrow * 1024 + ko + klane;
    const __bf16* aptr = &Abuf[buf][mrow][klane];
#pragma unroll 4
    for (int k = 0; k < CHK; k += 32) {
      v16bf av = *(const v16bf*)(aptr + k);
      v16bf bv = *(const v16bf*)(wptr + k);
      __builtin_prefetch(wptr + k + 512, 0, 1);   // global_prefetch_b8
      acc = __builtin_amdgcn_wmma_f32_16x16x32_bf16(
          false, av, false, bv, (short)0, acc, false, false);
    }
    return acc;
  };

  // End-of-chunk handshake: all waves done reading current buffer; wave 0
  // drains the in-flight TDM; second barrier publishes the next chunk.
  auto chunkSync = [&](bool more) {
    __syncthreads();
    if (wave == 0 && more) __builtin_amdgcn_s_wait_tensorcnt(0);
    __syncthreads();
  };

  // One layer's GEMM: gates = srcA @ Wih^T + srcH @ Whh^T (K = 1024 each),
  // staged as 4 straight-line TDM chunks of 16x512 bf16, double-buffered.
  auto layerGemm = [&](const __bf16* __restrict__ srcA,
                       const __bf16* __restrict__ Wih,
                       const __bf16* __restrict__ srcH,
                       const __bf16* __restrict__ Whh) -> v8f {
    v8f acc = {};
    if (wave == 0) {                       // prologue: chunk 0 -> buf 0
      tdm_load_tile(ldsA0, srcA, 16, CHK, 1024);
      __builtin_amdgcn_s_wait_tensorcnt(0);
    }
    __syncthreads();

    if (wave == 0) tdm_load_tile(ldsA1, srcA + CHK, 16, CHK, 1024);
    acc = gemmChunk(Wih, 0, 0, acc);
    chunkSync(true);

    if (wave == 0) tdm_load_tile(ldsA0, srcH, 16, CHK, 1024);
    acc = gemmChunk(Wih, CHK, 1, acc);
    chunkSync(true);

    if (wave == 0) tdm_load_tile(ldsA1, srcH + CHK, 16, CHK, 1024);
    acc = gemmChunk(Whh, 0, 0, acc);
    chunkSync(true);

    acc = gemmChunk(Whh, CHK, 1, acc);
    __syncthreads();
    return acc;
  };

  // Stage accumulator tile into gbuf (C layout: VGPR r -> M = r + 8*(lane>=16)).
  auto stageGates = [&](v8f acc) {
    int mb = (lane >> 4) << 3;
#pragma unroll
    for (int r = 0; r < 8; ++r)
      gbuf[gate][mb + r][tile * 16 + mrow] = acc[r];
  };

  // Sense-reversing agent-scope barrier across the 32 persistent WGs.
  auto gbarrier = [&]() {
    __threadfence();
    __syncthreads();
    if (threadIdx.x == 0) {
      int s = lsense ^ 1;
      lsense = s;
      if (__hip_atomic_fetch_add(&bar[0], 1, __ATOMIC_ACQ_REL,
                                 __HIP_MEMORY_SCOPE_AGENT) == NWG - 1) {
        __hip_atomic_store(&bar[0], 0, __ATOMIC_RELAXED, __HIP_MEMORY_SCOPE_AGENT);
        __hip_atomic_store(&bar[1], s, __ATOMIC_RELEASE, __HIP_MEMORY_SCOPE_AGENT);
      } else {
        while (__hip_atomic_load(&bar[1], __ATOMIC_ACQUIRE,
                                 __HIP_MEMORY_SCOPE_AGENT) != s)
          __builtin_amdgcn_s_sleep(2);
      }
    }
    __syncthreads();
    __threadfence();
  };

  // Gate nonlinearities + state update for this WG's 32 units x 16 batches.
  auto pointwise = [&](const float* __restrict__ bias, float (&cst)[16][32],
                       __bf16* __restrict__ hW, float* __restrict__ outT,
                       float* __restrict__ hFin, float* __restrict__ cFin,
                       bool last) {
    for (int p = threadIdx.x; p < 512; p += THREADS) {
      int bb = p >> 5, u = p & 31;
      int uG = uBase + u;
      float iv = gbuf[0][bb][u] + bias[uG];
      float fv = gbuf[1][bb][u] + bias[1024 + uG];
      float gv = gbuf[2][bb][u] + bias[2048 + uG];
      float ov = gbuf[3][bb][u] + bias[3072 + uG];
      float ig = expf(iv);                 // exponential input gate
      float fg = expf(fv);                 // exponential forget gate
      float gg = tanhf(gv);
      float og = 1.0f / (1.0f + expf(-ov));
      float c  = fg * cst[bb][u] + ig * gg;
      cst[bb][u] = c;
      float h = og * tanhf(c);
      hW[(size_t)bb * HH + uG] = (__bf16)h;          // bf16 state for GEMMs
      if (outT) outT[(size_t)bb * TT * HH + uG] = h; // f32 sequence output
      if (last) {
        hFin[(size_t)bb * HH + uG] = h;
        cFin[(size_t)bb * HH + uG] = c;
      }
    }
  };

#pragma unroll 1
  for (int t = 0; t < TT; ++t) {
    const int wpar = t & 1;            // write-parity h buffer
    const int rpar = wpar ^ 1;         // read-parity (previous step's h)
    const bool last = (t == TT - 1);

    // ------- layer 0: gates = x_t @ Wih0^T + h0_prev @ Whh0^T -------------
    v8f acc = layerGemm(xb + (size_t)t * BH, wih0,
                        h0buf + (size_t)rpar * BH, whh0);
    stageGates(acc);
    __syncthreads();
    pointwise(b0, c0s, h0buf + (size_t)wpar * BH, nullptr, h0f, c0f, last);
    gbarrier();                        // h0(t) visible to all WGs

    // ------- layer 1: gates = h0_t @ Wih1^T + h1_prev @ Whh1^T ------------
    acc = layerGemm(h0buf + (size_t)wpar * BH, wih1,
                    h1buf + (size_t)rpar * BH, whh1);
    stageGates(acc);
    __syncthreads();
    pointwise(b1, c1s, h1buf + (size_t)wpar * BH,
              out + (size_t)t * HH, h1f, c1f, last);
    gbarrier();                        // h1(t) visible to all WGs
  }
}

// ---------------------------------------------------------------------------
extern "C" void kernel_launch(void* const* d_in, const int* in_sizes, int n_in,
                              void* d_out, int out_size, void* d_ws, size_t ws_size,
                              hipStream_t stream) {
  (void)in_sizes; (void)n_in; (void)out_size; (void)ws_size;

  const float* x    = (const float*)d_in[0];
  const float* wih0 = (const float*)d_in[1];
  const float* whh0 = (const float*)d_in[2];
  const float* b0   = (const float*)d_in[3];
  const float* wih1 = (const float*)d_in[4];
  const float* whh1 = (const float*)d_in[5];
  const float* b1   = (const float*)d_in[6];

  // ---- workspace layout ----
  char* ws = (char*)d_ws;
  int* bar = (int*)ws;                                       // [count, sense]
  unsigned short* wb = (unsigned short*)(ws + 256);          // 4 bf16 matrices
  unsigned short* wih0b = wb;
  unsigned short* whh0b = wb + (size_t)WMAT;
  unsigned short* wih1b = wb + 2 * (size_t)WMAT;
  unsigned short* whh1b = wb + 3 * (size_t)WMAT;
  unsigned short* xb16  = wb + 4 * (size_t)WMAT;             // [T][B][D] bf16
  unsigned short* h0buf = xb16 + (size_t)BSZ * TT * DD;      // [2][B*H] bf16
  unsigned short* h1buf = h0buf + 2 * (size_t)BH;

  // ---- output layout: (out[B,T,H], h0, c0, h1, c1) concatenated ----
  float* out = (float*)d_out;
  float* h0f = out + (size_t)BSZ * TT * HH;
  float* c0f = h0f + BH;
  float* h1f = c0f + BH;
  float* c1f = h1f + BH;

  // 1) weights f32 -> bf16 (32 MB, L2-resident thereafter)
  convert_weights<<<(4 * WMAT) / THREADS, THREADS, 0, stream>>>(
      wih0, whh0, wih1, whh1, wb);

  // 2) x f32 [B,T,D] -> bf16 time-major [T,B,D] (per-step contiguous tiles)
  convert_x<<<((unsigned)((size_t)BSZ * TT * DD)) / THREADS, THREADS, 0, stream>>>(
      x, xb16);

  // 3) zero barrier state + both parities of h0/h1 bf16 buffers
  init_state<<<(4 * BH + 2 + THREADS - 1) / THREADS, THREADS, 0, stream>>>(
      bar, h0buf, 4 * BH);

  // 4) persistent recurrent kernel: 32 WGs x 256 threads (8 wave32 each)
  slstm_persistent<<<NWG, THREADS, 0, stream>>>(
      b0, b1, xb16, wih0b, whh0b, wih1b, whh1b,
      h0buf, h1buf, bar, out, h0f, c0f, h1f, c1f);
}
